// GPTDecoder_17506286698865
// MI455X (gfx1250) — matrix operand order repaired and checked by reading, not compiled, not run on hardware
//
#include <hip/hip_runtime.h>
#include <math.h>

typedef __attribute__((ext_vector_type(16))) __bf16       bf16x16;
typedef __attribute__((ext_vector_type(8)))  float        f32x8;
typedef __attribute__((ext_vector_type(4)))  unsigned int u32x4;

#define BS 256
#define NEG_MIN (-3.402823466e+38f)
#define EPSV 1e-12f

union ABTile { u32x4 q[2]; bf16x16 v; };

__device__ __forceinline__ unsigned short f2bf(float f) {
    unsigned int x = __float_as_uint(f);
    unsigned int r = x + 0x7fffu + ((x >> 16) & 1u);   // round-to-nearest-even
    return (unsigned short)(r >> 16);
}

// ---------------- setup kernels (run once) ----------------

__global__ void k_starts(const int* __restrict__ lens, int* __restrict__ starts, int B) {
    if (threadIdx.x == 0 && blockIdx.x == 0) {
        int s = 0;
        for (int b = 0; b < B; ++b) { starts[b] = s; s += lens[b]; }
    }
}

// enc[b,l,:] = (l < len[b]) ? h[start[b]+l, :] : 0   (written straight into d_out enc region)
__global__ void k_encbuild(const float* __restrict__ h, const int* __restrict__ lens,
                           const int* __restrict__ starts, float* __restrict__ enc,
                           int L, int H) {
    int row = blockIdx.x;               // b*L + l
    int b = row / L, l = row - b * L;
    float* dst = enc + (size_t)row * H;
    if (l < lens[b]) {
        const float* src = h + ((size_t)starts[b] + l) * H;
        for (int k = threadIdx.x; k < H; k += blockDim.x) dst[k] = src[k];
    } else {
        for (int k = threadIdx.x; k < H; k += blockDim.x) dst[k] = 0.0f;
    }
}

// enc_proj[row,u] = dot(enc[row,:], W1[u,:])   one wave32 per row
__global__ void k_encproj(const float* __restrict__ enc, const float* __restrict__ W1,
                          float* __restrict__ ep, int nrows, int H, int U) {
    int gid = blockIdx.x * blockDim.x + threadIdx.x;
    int row = gid >> 5, lane = gid & 31;
    if (row >= nrows) return;
    float acc[16];
    for (int u = 0; u < U; ++u) acc[u] = 0.f;
    const float* er = enc + (size_t)row * H;
    for (int k = lane; k < H; k += 32) {
        float e = er[k];
        for (int u = 0; u < U; ++u) acc[u] += e * W1[(size_t)u * H + k];
    }
    for (int u = 0; u < U; ++u) {
        float v = acc[u];
        for (int off = 16; off > 0; off >>= 1) v += __shfl_xor(v, off);
        if (lane == 0) ep[(size_t)row * U + u] = v;
    }
}

// zero cov/closs; dec_h = hg (f32 + bf16 image); xb (bf16 dec_in) = 0
__global__ void k_init(const float* __restrict__ hg, float* __restrict__ dec_h,
                       unsigned short* __restrict__ xb, unsigned short* __restrict__ hb,
                       float* __restrict__ cov, float* __restrict__ closs,
                       int BL, int BH, int B) {
    int i = blockIdx.x * blockDim.x + threadIdx.x;
    if (i < BL) cov[i] = 0.f;
    if (i < BH) {
        float v = hg[i];
        dec_h[i] = v;
        hb[i] = f2bf(v);
        xb[i] = 0;           // bf16(0.0f)
    }
    if (i < B) closs[i] = 0.f;
}

__global__ void k_wconv(const float* __restrict__ wih, const float* __restrict__ whh,
                        unsigned short* __restrict__ wib, unsigned short* __restrict__ whb, int n) {
    int i = blockIdx.x * blockDim.x + threadIdx.x;
    if (i < n) { wib[i] = f2bf(wih[i]); whb[i] = f2bf(whh[i]); }
}

// ---------------- per-step kernels ----------------

// G[m,n] = sum_k X[m,k]*W[n,k]  via V_WMMA_F32_16X16X32_BF16 (bf16 in, f32 acc)
// blockIdx.z: 0 -> (xb, wib, gi),  1 -> (hb, whb, gh)
__global__ void __launch_bounds__(BS) k_gemm(
        const unsigned short* __restrict__ xb, const unsigned short* __restrict__ hb,
        const unsigned short* __restrict__ wib, const unsigned short* __restrict__ whb,
        float* __restrict__ gi, float* __restrict__ gh, int B, int H, int N3) {
    const unsigned short* X = blockIdx.z ? hb : xb;
    const unsigned short* W = blockIdx.z ? whb : wib;
    float* G = blockIdx.z ? gh : gi;

    int tid  = threadIdx.x;
    int wave = tid >> 5, lane = tid & 31;
    int half = lane >> 4, lm = lane & 15;
    int m0 = blockIdx.y * 32 + (wave & 1) * 16;
    int n0 = blockIdx.x * 64 + (wave >> 1) * 16;
    int mrow = m0 + lm; if (mrow >= B)  mrow = B - 1;
    int nrow = n0 + lm; if (nrow >= N3) nrow = N3 - 1;
    const unsigned short* xr = X + (size_t)mrow * H;   // A: row m, 16-bit A layout
    const unsigned short* wr = W + (size_t)nrow * H;   // B: column n of W^T = row n of W

    f32x8 acc = {0.f, 0.f, 0.f, 0.f, 0.f, 0.f, 0.f, 0.f};
    for (int k0 = 0; k0 < H; k0 += 32) {
        ABTile a, b;
        // A 16x32 bf16: lane half=0 holds K k0+0..7 & k0+16..23; half=1 holds +8..15 & +24..31
        a.q[0] = *(const u32x4*)(xr + k0 + half * 8);
        a.q[1] = *(const u32x4*)(xr + k0 + 16 + half * 8);
        // B 32x16 bf16: lane half=0 holds K k0+0..7 & k0+16..23 of col lm; half=1 holds +8..15 & +24..31
        b.q[0] = *(const u32x4*)(wr + k0 + half * 8);
        b.q[1] = *(const u32x4*)(wr + k0 + 16 + half * 8);
        acc = __builtin_amdgcn_wmma_f32_16x16x32_bf16(false, a.v, false, b.v,
                                                      (short)0, acc, false, false);
    }
    // C/D layout: VGPR r -> M = r + 8*half, N = lm
    int ncol = n0 + lm;
    if (ncol < N3) {
        int mbase = m0 + half * 8;
#pragma unroll
        for (int r = 0; r < 8; ++r) {
            int m = mbase + r;
            if (m < B) G[(size_t)m * N3 + ncol] = acc[r];
        }
    }
}

// torch GRUCell gates (biases added here; GEMM omitted them); also emit bf16 dec_h
__global__ void k_gate(const float* __restrict__ gi, const float* __restrict__ gh,
                       const float* __restrict__ bih, const float* __restrict__ bhh,
                       float* __restrict__ dec_h, unsigned short* __restrict__ hb,
                       int B, int H) {
    int i = blockIdx.x * blockDim.x + threadIdx.x;
    if (i >= B * H) return;
    int b = i / H, k = i - b * H;
    size_t o = (size_t)b * 3 * H;
    float ir = gi[o + k]         + bih[k];
    float iz = gi[o + H + k]     + bih[H + k];
    float in = gi[o + 2 * H + k] + bih[2 * H + k];
    float hr = gh[o + k]         + bhh[k];
    float hz = gh[o + H + k]     + bhh[H + k];
    float hn = gh[o + 2 * H + k] + bhh[2 * H + k];
    float r = 1.f / (1.f + expf(-(ir + hr)));
    float z = 1.f / (1.f + expf(-(iz + hz)));
    float n = tanhf(in + r * hn);
    float hv = (1.f - z) * n + z * dec_h[i];
    dec_h[i] = hv;
    hb[i] = f2bf(hv);
}

// One block per batch row: attention + coverage + gumbel-softmax + pointer write
// + dec_in gather (emitted directly as bf16 GEMM input xb)
__global__ void __launch_bounds__(BS) k_attn(
        const float* __restrict__ ep, const float* __restrict__ enc,
        float* __restrict__ cov, const float* __restrict__ dec_h,
        unsigned short* __restrict__ xb,
        const float* __restrict__ W2, const float* __restrict__ wc,
        const float* __restrict__ V, const float* __restrict__ gum,
        const int* __restrict__ lens,
        float* __restrict__ yout, float* __restrict__ closs,
        int B, int L, int H, int U, int T, int t) {
    extern __shared__ float dyn[];
    float* ajb = dyn;          // [L]
    float* zb  = dyn + L;      // [L]
    __shared__ float s_u[16];
    __shared__ float sred[BS];
    __shared__ int   sidx[BS];

    int tid = threadIdx.x;
    int b = blockIdx.x;
    int len = lens[b];
    int wave = tid >> 5, lane = tid & 31;

    // s[u] = dot(dec_h[b,:], W2[u,:])
    const float* dh = dec_h + (size_t)b * H;
    for (int u = wave; u < U; u += (BS >> 5)) {
        float acc = 0.f;
        for (int k = lane; k < H; k += 32) acc += dh[k] * W2[(size_t)u * H + k];
        for (int off = 16; off > 0; off >>= 1) acc += __shfl_xor(acc, off);
        if (lane == 0) s_u[u] = acc;
    }
    __syncthreads();

    // uj with mask
    const float* eprow = ep + (size_t)b * L * U;
    float* covr = cov + (size_t)b * L;
    float lmax = NEG_MIN;
    for (int l = tid; l < L; l += BS) {
        float cv = covr[l];
        const float* e = eprow + (size_t)l * U;
        float a = 0.f;
        for (int u = 0; u < U; ++u) a += V[u] * tanhf(e[u] + s_u[u] + cv * wc[u]);
        float uv = (l < len) ? a : NEG_MIN;
        ajb[l] = uv;
        lmax = fmaxf(lmax, uv);
    }
    sred[tid] = lmax; __syncthreads();
    for (int s = BS >> 1; s > 0; s >>= 1) { if (tid < s) sred[tid] = fmaxf(sred[tid], sred[tid + s]); __syncthreads(); }
    float M1 = sred[0]; __syncthreads();

    float lsum = 0.f;
    for (int l = tid; l < L; l += BS) { float e2 = expf(ajb[l] - M1); ajb[l] = e2; lsum += e2; }
    sred[tid] = lsum; __syncthreads();
    for (int s = BS >> 1; s > 0; s >>= 1) { if (tid < s) sred[tid] += sred[tid + s]; __syncthreads(); }
    float inv = 1.f / sred[0]; __syncthreads();

    // aj, coverage loss (pre-update cov), cov += aj
    float lcl = 0.f;
    for (int l = tid; l < L; l += BS) {
        float a = ajb[l] * inv; ajb[l] = a;
        float cv = covr[l];
        lcl += fminf(a, cv);
        covr[l] = cv + a;
    }
    sred[tid] = lcl; __syncthreads();
    for (int s = BS >> 1; s > 0; s >>= 1) { if (tid < s) sred[tid] += sred[tid + s]; __syncthreads(); }
    if (tid == 0) closs[b] += sred[0];
    __syncthreads();

    // gumbel-softmax logits (no mask, matching reference)
    const float* gr = gum + ((size_t)t * B + b) * L;
    const float inv_tau = 1.f / 3.0f;
    float lmax2 = NEG_MIN;
    for (int l = tid; l < L; l += BS) {
        float g = -logf(-logf(gr[l] + EPSV) + EPSV);
        float zv = (logf(ajb[l] + EPSV) + g) * inv_tau;
        zb[l] = zv;
        lmax2 = fmaxf(lmax2, zv);
    }
    sred[tid] = lmax2; __syncthreads();
    for (int s = BS >> 1; s > 0; s >>= 1) { if (tid < s) sred[tid] = fmaxf(sred[tid], sred[tid + s]); __syncthreads(); }
    float M2 = sred[0]; __syncthreads();

    float lsum2 = 0.f;
    for (int l = tid; l < L; l += BS) lsum2 += expf(zb[l] - M2);
    sred[tid] = lsum2; __syncthreads();
    for (int s = BS >> 1; s > 0; s >>= 1) { if (tid < s) sred[tid] += sred[tid + s]; __syncthreads(); }
    float S2 = sred[0]; __syncthreads();

    // argmax of y_soft == argmax of z (tie -> lowest index, like jnp.argmax)
    float bv = NEG_MIN; int bi = L;
    for (int l = tid; l < L; l += BS) {
        float zv = zb[l];
        if (zv > bv || (zv == bv && l < bi)) { bv = zv; bi = l; }
    }
    sred[tid] = bv; sidx[tid] = bi; __syncthreads();
    for (int s = BS >> 1; s > 0; s >>= 1) {
        if (tid < s) {
            float ov = sred[tid + s]; int oi = sidx[tid + s];
            if (ov > sred[tid] || (ov == sred[tid] && oi < sidx[tid])) { sred[tid] = ov; sidx[tid] = oi; }
        }
        __syncthreads();
    }
    int lstar = sidx[0];
    float ys = expf(zb[lstar] - M2) / S2;
    float vstar = (1.0f - ys) + ys;  // straight-through value at argmax; exactly 0 elsewhere

    // pointer_argmaxs[b,l,t]
    float* yo = yout + (size_t)b * L * T + t;
    for (int l = tid; l < L; l += BS) yo[(size_t)l * T] = (l == lstar) ? vstar : 0.0f;

    // next GEMM input: dec_in[b,:] = vstar * enc[b,lstar,:], stored as bf16 directly
    const float* erow = enc + ((size_t)b * L + lstar) * H;
    unsigned short* di = xb + (size_t)b * H;
    for (int k = tid; k < H; k += BS) di[k] = f2bf(vstar * erow[k]);
}

__global__ void k_final(const float* __restrict__ closs, float* __restrict__ out, int B) {
    if (threadIdx.x == 0 && blockIdx.x == 0) {
        float s = 0.f;
        for (int b = 0; b < B; ++b) s += closs[b];
        out[0] = s;
    }
}

// ---------------- launch ----------------

extern "C" void kernel_launch(void* const* d_in, const int* in_sizes, int n_in,
                              void* d_out, int out_size, void* d_ws, size_t ws_size,
                              hipStream_t stream) {
    const float* h   = (const float*)d_in[0];
    const float* hg  = (const float*)d_in[1];
    const int*   lens= (const int*)  d_in[2];
    const float* gum = (const float*)d_in[3];
    const float* W1  = (const float*)d_in[4];
    const float* W2  = (const float*)d_in[5];
    const float* wc  = (const float*)d_in[6];
    const float* V   = (const float*)d_in[7];
    const float* wih = (const float*)d_in[8];
    const float* whh = (const float*)d_in[9];
    const float* bih = (const float*)d_in[10];
    const float* bhh = (const float*)d_in[11];

    int U = in_sizes[6];                 // wc [U,1]
    if (U > 16) U = 16;
    int H = in_sizes[4] / U;             // W1 [U,H]
    int B = in_sizes[1] / H;             // hg [B,H]
    int L = (int)((long long)in_sizes[0] / ((long long)B * H));   // h [B*L,H]
    int T = (int)((long long)in_sizes[3] / ((long long)B * L));   // gumbel_u [T,B,L]
    int N3 = 3 * H;
    int BH = B * H, BL = B * L;

    float* out0 = (float*)d_out;                 // pointer_argmaxs [B,L,T]
    float* out1 = out0 + (size_t)B * L * T;      // cov_loss scalar
    float* out2 = out1 + 1;                      // enc [B,L,H]

    float* ws = (float*)d_ws;
    size_t off = 0;
    auto alloc = [&](size_t n) { size_t o = off; off += (n + 63) & ~(size_t)63; return o; };
    int*            starts  = (int*)(ws + alloc(B));
    float*          encproj = ws + alloc((size_t)BL * U);
    float*          cov     = ws + alloc((size_t)BL);
    float*          dec_h   = ws + alloc((size_t)BH);
    float*          gi      = ws + alloc((size_t)B * N3);
    float*          gh      = ws + alloc((size_t)B * N3);
    unsigned short* xb      = (unsigned short*)(ws + alloc(((size_t)BH + 1) / 2));
    unsigned short* hb      = (unsigned short*)(ws + alloc(((size_t)BH + 1) / 2));
    unsigned short* wib     = (unsigned short*)(ws + alloc(((size_t)N3 * H + 1) / 2));
    unsigned short* whb     = (unsigned short*)(ws + alloc(((size_t)N3 * H + 1) / 2));
    float*          closs   = ws + alloc(B);
    (void)ws_size; (void)n_in; (void)out_size;

    // one-time setup
    k_starts<<<1, 1, 0, stream>>>(lens, starts, B);
    k_encbuild<<<BL, 256, 0, stream>>>(h, lens, starts, out2, L, H);
    {
        long long thr = (long long)BL * 32;        // one wave per (b,l)
        k_encproj<<<(unsigned)((thr + 255) / 256), 256, 0, stream>>>(out2, W1, encproj, BL, H, U);
    }
    {
        int mx = BL > BH ? BL : BH;
        k_init<<<(mx + 255) / 256, 256, 0, stream>>>(hg, dec_h, xb, hb, cov, closs, BL, BH, B);
    }
    k_wconv<<<(N3 * H + 255) / 256, 256, 0, stream>>>(wih, whh, wib, whb, N3 * H);

    // sequential decode: 3 kernels per step on the dependency chain
    dim3 gg((N3 + 63) / 64, (B + 31) / 32, 2);
    for (int t = 0; t < T; ++t) {
        k_gemm<<<gg, BS, 0, stream>>>(xb, hb, wib, whb, gi, gh, B, H, N3);
        k_gate<<<(BH + 255) / 256, 256, 0, stream>>>(gi, gh, bih, bhh, dec_h, hb, B, H);
        k_attn<<<B, BS, 2 * L * (int)sizeof(float), stream>>>(
            encproj, out2, cov, dec_h, xb, W2, wc, V, gum, lens,
            out0, closs, B, L, H, U, T, t);
    }
    k_final<<<1, 1, 0, stream>>>(closs, out1, B);
}
